// graphAttention_38534446579798
// MI455X (gfx1250) — matrix operand order repaired and checked by reading, not compiled, not run on hardware
//
#include <hip/hip_runtime.h>
#include <hip/hip_bf16.h>

typedef __attribute__((ext_vector_type(16))) _Float16 v16h;
typedef __attribute__((ext_vector_type(8)))  float    v8f;

#define NATOM 10000
#define NEDGE 40000
#define DD 64
#define BDIM 12
#define BN_EPS_F 1e-6f
#define SM_EPS_F 1e-8f

__device__ __forceinline__ v8f wmma_f16(v16h a, v16h b, v8f c) {
  // D = A(16x32 f16) x B(32x16 f16) + C(16x16 f32)
  return __builtin_amdgcn_wmma_f32_16x16x32_f16(false, a, false, b, (short)0, c, false, false);
}

// monotone float<->int mapping for atomicMax on signed ints
__device__ __forceinline__ int f2ord(float f) {
  int i = __float_as_int(f);
  return (i >= 0) ? i : (i ^ 0x7fffffff);
}
__device__ __forceinline__ float ord2f(int o) {
  int i = (o >= 0) ? o : (o ^ 0x7fffffff);
  return __int_as_float(i);
}

// ---------------------------------------------------------------------------
// K0: init workspace; fold BN(+bias) into per-column scale/shift; enc_W -> f16
// ---------------------------------------------------------------------------
__global__ void k0_init(const float* __restrict__ enc_W, const float* __restrict__ enc_b,
                        const float* __restrict__ enc_gamma, const float* __restrict__ enc_beta,
                        const float* __restrict__ enc_mean, const float* __restrict__ enc_var,
                        const float* __restrict__ att_b, const float* __restrict__ att_gamma,
                        const float* __restrict__ att_beta, const float* __restrict__ att_mean,
                        const float* __restrict__ att_var,
                        float* __restrict__ ctx, int* __restrict__ mx_ord,
                        float* __restrict__ denom,
                        float* __restrict__ enc_scale, float* __restrict__ enc_shift,
                        float* __restrict__ att_scale, float* __restrict__ att_shift,
                        _Float16* __restrict__ ewh)
{
  int idx = blockIdx.x * blockDim.x + threadIdx.x;
  if (idx < NATOM * DD) ctx[idx] = 0.f;
  if (idx < NATOM) { mx_ord[idx] = (int)0x80000000; denom[idx] = 0.f; }
  if (idx < DD * DD) {
    float inv = rsqrtf(enc_var[idx] + BN_EPS_F);
    float sc  = enc_gamma[idx] * inv;
    enc_scale[idx] = sc;
    enc_shift[idx] = (enc_b[idx] - enc_mean[idx]) * sc + enc_beta[idx];
  }
  if (idx < DD) {
    float inv = rsqrtf(att_var[idx] + BN_EPS_F);
    float sc  = att_gamma[idx] * inv;
    att_scale[idx] = sc;
    att_shift[idx] = (att_b[idx] - att_mean[idx]) * sc + att_beta[idx];
  }
  if (idx < DD * DD * 16) {            // enc_W rows padded K 12 -> 16, f16
    int r = idx >> 4, k = idx & 15;
    ewh[idx] = (k < BDIM) ? (_Float16)enc_W[r * BDIM + k] : (_Float16)0.f;
  }
}

// ---------------------------------------------------------------------------
// K1: fused enc-GEMM (WMMA) + BN/ReLU + d-contraction -> neighbor[E,64]
// 16 edges per block, 4 waves (N-tile of 16 cols per wave), loop d=0..63
// ---------------------------------------------------------------------------
__global__ void __launch_bounds__(128) k1_neighbor(
    const float* __restrict__ atom, const int* __restrict__ bond_index,
    const float* __restrict__ bond, const _Float16* __restrict__ ewh,
    const float* __restrict__ enc_scale, const float* __restrict__ enc_shift,
    float* __restrict__ neighbor)
{
  __shared__ float sAtom[16][DD];
  __shared__ int   sNbr[16];
  const int tid  = threadIdx.x;
  const int wave = tid >> 5, lane = tid & 31;
  const int h = lane >> 4, nl = lane & 15;
  const int e_base = blockIdx.x * 16;

  if (tid < 16) sNbr[tid] = bond_index[(e_base + tid) * 2 + 1];
  __syncthreads();
  for (int i = tid; i < 16 * DD; i += 128) {
    int el = i >> 6, d = i & 63;
    sAtom[el][d] = atom[sNbr[el] * DD + d];
  }
  __syncthreads();

  // A fragment: bond tile, 16 edges x K32 (real K=12, rest zero). lane&15 = M.
  v16h afrag;
  {
    const int e = e_base + nl;
    for (int v = 0; v < 8; ++v) {
      int kb = h * 16 + 2 * v;
      for (int j = 0; j < 2; ++j) {
        int k = kb + j;
        afrag[2 * v + j] = (k < BDIM) ? (_Float16)bond[e * BDIM + k] : (_Float16)0.f;
      }
    }
  }

  v8f acc; for (int i = 0; i < 8; ++i) acc[i] = 0.f;
  const int fcol = wave * 16 + nl;   // this lane's output column (0..63)

  for (int d = 0; d < DD; ++d) {
    // B fragment: enc_W^T slice, K x N = 32x16; K>=16 is zero-padded
    v16h bfrag;
    if (h == 0) {
      const _Float16* p = ewh + (d * DD + fcol) * 16;
      for (int t = 0; t < 16; ++t) bfrag[t] = p[t];
    } else {
      for (int t = 0; t < 16; ++t) bfrag[t] = (_Float16)0.f;
    }
    v8f c; for (int i = 0; i < 8; ++i) c[i] = 0.f;
    c = wmma_f16(afrag, bfrag, c);

    const float sc = enc_scale[d * DD + fcol];
    const float sh = enc_shift[d * DD + fcol];
    for (int i = 0; i < 8; ++i) {
      float x = fmaf(c[i], sc, sh);       // BN (bias folded)
      x = fmaxf(x, 0.f);                  // ReLU
      acc[i] = fmaf(x, sAtom[h * 8 + i][d], acc[i]);  // contract over d
    }
  }
  for (int i = 0; i < 8; ++i) {
    int e = e_base + h * 8 + i;
    neighbor[e * DD + fcol] = acc[i];
  }
}

// ---------------------------------------------------------------------------
// K2: attention score per edge (wave-per-edge, shuffle reduce) + segment max
// ---------------------------------------------------------------------------
__global__ void __launch_bounds__(256) k2_score(
    const float* __restrict__ atom, const int* __restrict__ bond_index,
    const float* __restrict__ neighbor, const float* __restrict__ align_W,
    const float* __restrict__ align_b,
    float* __restrict__ score, int* __restrict__ mx_ord)
{
  const int wave = threadIdx.x >> 5, lane = threadIdx.x & 31;
  const int e = blockIdx.x * 8 + wave;
  if (e >= NEDGE) return;
  const int tgt = bond_index[e * 2];
  float v = atom[tgt * DD + lane]        * align_W[lane]
          + atom[tgt * DD + lane + 32]   * align_W[lane + 32]
          + neighbor[e * DD + lane]      * align_W[DD + lane]
          + neighbor[e * DD + lane + 32] * align_W[96 + lane];
  for (int off = 16; off >= 1; off >>= 1) v += __shfl_xor(v, off, 32);
  if (lane == 0) {
    float s = v + align_b[0];
    s = (s > 0.f) ? s : 0.01f * s;       // leaky_relu
    score[e] = s;
    atomicMax(&mx_ord[tgt], f2ord(s));
  }
}

// ---------------------------------------------------------------------------
// K3: ex = exp(score - mx[tgt]); denom = segment_sum(ex)
// ---------------------------------------------------------------------------
__global__ void k3_expsum(const int* __restrict__ bond_index,
                          const int* __restrict__ mx_ord,
                          float* __restrict__ score_ex, float* __restrict__ denom)
{
  const int e = blockIdx.x * blockDim.x + threadIdx.x;
  if (e >= NEDGE) return;
  const int tgt = bond_index[e * 2];
  const float m  = ord2f(mx_ord[tgt]);
  const float ex = expf(score_ex[e] - m);
  score_ex[e] = ex;
  atomicAdd(&denom[tgt], ex);
}

// ---------------------------------------------------------------------------
// K4: attd = BN(neighbor @ att_W.T) via WMMA; context += attn * attd
// ---------------------------------------------------------------------------
__global__ void __launch_bounds__(128) k4_context(
    const int* __restrict__ bond_index, const float* __restrict__ neighbor,
    const float* __restrict__ att_W, const float* __restrict__ att_scale,
    const float* __restrict__ att_shift, const float* __restrict__ exf,
    const float* __restrict__ denom, float* __restrict__ ctx)
{
  __shared__ int   sTgt[16];
  __shared__ float sAttn[16];
  const int tid = threadIdx.x;
  const int wave = tid >> 5, lane = tid & 31, h = lane >> 4, nl = lane & 15;
  const int e_base = blockIdx.x * 16;
  if (tid < 16) {
    const int e = e_base + tid;
    const int tgt = bond_index[e * 2];
    sTgt[tid]  = tgt;
    sAttn[tid] = exf[e] / (denom[tgt] + SM_EPS_F);
  }
  __syncthreads();

  // A: neighbor tile 16x64 -> two 16x32 f16 fragments
  v16h a0, a1;
  {
    const float* row = neighbor + (e_base + nl) * DD;
    for (int v = 0; v < 8; ++v) {
      int kb = h * 16 + 2 * v;
      a0[2*v]   = (_Float16)row[kb];      a0[2*v+1] = (_Float16)row[kb + 1];
      a1[2*v]   = (_Float16)row[kb + 32]; a1[2*v+1] = (_Float16)row[kb + 33];
    }
  }
  const int cb = wave * 16 + nl;         // output column (0..63)
  v16h b0, b1;
  {
    const float* wr = att_W + cb * DD;   // B[k][n] = att_W[n_col, k]
    for (int v = 0; v < 8; ++v) {
      int k = h * 16 + 2 * v;
      b0[2*v]   = (_Float16)wr[k];      b0[2*v+1] = (_Float16)wr[k + 1];
      b1[2*v]   = (_Float16)wr[k + 32]; b1[2*v+1] = (_Float16)wr[k + 33];
    }
  }
  v8f c; for (int i = 0; i < 8; ++i) c[i] = 0.f;
  c = wmma_f16(a0, b0, c);
  c = wmma_f16(a1, b1, c);

  const float sc = att_scale[cb], sh = att_shift[cb];
  for (int i = 0; i < 8; ++i) {
    int row = h * 8 + i;
    float attd = fmaf(c[i], sc, sh);
    atomicAdd(&ctx[sTgt[row] * DD + cb], sAttn[row] * attd);
  }
}

// ---------------------------------------------------------------------------
// K5: GRU cell. Per block: 16 atoms; per wave: 16 output cols x 3 gates,
// gi = elu(ctx) @ W_ih.T, gh = atom @ W_hh.T (WMMA, K=64)
// ---------------------------------------------------------------------------
__global__ void __launch_bounds__(128) k5_gru(
    const float* __restrict__ atom, const float* __restrict__ ctx,
    const float* __restrict__ W_ih, const float* __restrict__ W_hh,
    const float* __restrict__ b_ih, const float* __restrict__ b_hh,
    float* __restrict__ out)
{
  const int tid = threadIdx.x;
  const int wave = tid >> 5, lane = tid & 31, h = lane >> 4, nl = lane & 15;
  const int a_base = blockIdx.x * 16;
  const int arow = a_base + nl;

  v16h ax0, ax1;  // elu(context)
  v16h ah0, ah1;  // atom (hidden state)
  for (int v = 0; v < 8; ++v) {
    int kb = h * 16 + 2 * v;
    for (int j = 0; j < 2; ++j) {
      int k = kb + j;
      float c0 = ctx[arow * DD + k];
      float c1 = ctx[arow * DD + k + 32];
      c0 = (c0 > 0.f) ? c0 : (expf(c0) - 1.f);
      c1 = (c1 > 0.f) ? c1 : (expf(c1) - 1.f);
      ax0[2*v+j] = (_Float16)c0;                    ax1[2*v+j] = (_Float16)c1;
      ah0[2*v+j] = (_Float16)atom[arow * DD + k];   ah1[2*v+j] = (_Float16)atom[arow * DD + k + 32];
    }
  }

  const int cl = wave * 16 + nl;        // output col within D
  float gi[3][8], gh[3][8];
  for (int g = 0; g < 3; ++g) {
    const int Ng = g * DD + cl;         // row of W (0..191)
    v16h b0, b1;
    const float* wr = W_ih + Ng * DD;
    for (int v = 0; v < 8; ++v) { int k = h * 16 + 2 * v;
      b0[2*v] = (_Float16)wr[k];      b0[2*v+1] = (_Float16)wr[k + 1];
      b1[2*v] = (_Float16)wr[k + 32]; b1[2*v+1] = (_Float16)wr[k + 33]; }
    v8f c; for (int i = 0; i < 8; ++i) c[i] = 0.f;
    c = wmma_f16(ax0, b0, c); c = wmma_f16(ax1, b1, c);
    float bi = b_ih[Ng];
    for (int i = 0; i < 8; ++i) gi[g][i] = c[i] + bi;

    const float* wh = W_hh + Ng * DD;
    for (int v = 0; v < 8; ++v) { int k = h * 16 + 2 * v;
      b0[2*v] = (_Float16)wh[k];      b0[2*v+1] = (_Float16)wh[k + 1];
      b1[2*v] = (_Float16)wh[k + 32]; b1[2*v+1] = (_Float16)wh[k + 33]; }
    v8f d; for (int i = 0; i < 8; ++i) d[i] = 0.f;
    d = wmma_f16(ah0, b0, d); d = wmma_f16(ah1, b1, d);
    float bh = b_hh[Ng];
    for (int i = 0; i < 8; ++i) gh[g][i] = d[i] + bh;
  }

  for (int i = 0; i < 8; ++i) {
    const int row = a_base + h * 8 + i;
    float r = 1.f / (1.f + expf(-(gi[0][i] + gh[0][i])));
    float z = 1.f / (1.f + expf(-(gi[1][i] + gh[1][i])));
    float n = tanhf(gi[2][i] + r * gh[2][i]);
    float hv = atom[row * DD + cl];
    out[row * DD + cl] = (1.f - z) * n + z * hv;
  }
}

// ---------------------------------------------------------------------------
extern "C" void kernel_launch(void* const* d_in, const int* in_sizes, int n_in,
                              void* d_out, int out_size, void* d_ws, size_t ws_size,
                              hipStream_t stream) {
  const float* atom       = (const float*)d_in[0];
  const int*   bond_index = (const int*)  d_in[1];
  const float* bond       = (const float*)d_in[2];
  const float* enc_W      = (const float*)d_in[3];
  const float* enc_b      = (const float*)d_in[4];
  const float* enc_gamma  = (const float*)d_in[5];
  const float* enc_beta   = (const float*)d_in[6];
  const float* enc_mean   = (const float*)d_in[7];
  const float* enc_var    = (const float*)d_in[8];
  const float* align_W    = (const float*)d_in[9];
  const float* align_b    = (const float*)d_in[10];
  const float* att_W      = (const float*)d_in[11];
  const float* att_b      = (const float*)d_in[12];
  const float* att_gamma  = (const float*)d_in[13];
  const float* att_beta   = (const float*)d_in[14];
  const float* att_mean   = (const float*)d_in[15];
  const float* att_var    = (const float*)d_in[16];
  const float* gru_W_ih   = (const float*)d_in[17];
  const float* gru_W_hh   = (const float*)d_in[18];
  const float* gru_b_ih   = (const float*)d_in[19];
  const float* gru_b_hh   = (const float*)d_in[20];

  float* ws        = (float*)d_ws;
  float* neighbor  = ws;                         // E*64
  float* exf       = neighbor + NEDGE * DD;      // E   (score, then exp)
  int*   mx_ord    = (int*)(exf + NEDGE);        // N_ATOM
  float* denom     = (float*)(mx_ord + NATOM);   // N_ATOM
  float* ctx       = denom + NATOM;              // N_ATOM*64
  float* enc_scale = ctx + NATOM * DD;           // 4096
  float* enc_shift = enc_scale + DD * DD;        // 4096
  float* att_scale = enc_shift + DD * DD;        // 64
  float* att_shift = att_scale + DD;             // 64
  _Float16* ewh    = (_Float16*)(att_shift + DD);// 4096*16 halves

  k0_init<<<2500, 256, 0, stream>>>(enc_W, enc_b, enc_gamma, enc_beta, enc_mean, enc_var,
                                    att_b, att_gamma, att_beta, att_mean, att_var,
                                    ctx, mx_ord, denom,
                                    enc_scale, enc_shift, att_scale, att_shift, ewh);
  k1_neighbor<<<NEDGE / 16, 128, 0, stream>>>(atom, bond_index, bond, ewh,
                                              enc_scale, enc_shift, neighbor);
  k2_score<<<NEDGE / 8, 256, 0, stream>>>(atom, bond_index, neighbor,
                                          align_W, align_b, exf, mx_ord);
  k3_expsum<<<(NEDGE + 255) / 256, 256, 0, stream>>>(bond_index, mx_ord, exf, denom);
  k4_context<<<NEDGE / 16, 128, 0, stream>>>(bond_index, neighbor, att_W,
                                             att_scale, att_shift, exf, denom, ctx);
  k5_gru<<<NATOM / 16, 128, 0, stream>>>(atom, ctx, gru_W_ih, gru_W_hh,
                                         gru_b_ih, gru_b_hh, (float*)d_out);
}
